// SlotAttention_85237920957033
// MI455X (gfx1250) — matrix operand order repaired and checked by reading, not compiled, not run on hardware
//
#include <hip/hip_runtime.h>
#include <hip/hip_bf16.h>

typedef __attribute__((ext_vector_type(16))) __bf16 v16bf;
typedef __attribute__((ext_vector_type(8)))  __bf16 v8bf;
typedef __attribute__((ext_vector_type(8)))  float  v8f;

#define WMMA_BF16(a, b, c) \
  __builtin_amdgcn_wmma_f32_16x16x32_bf16(false, (a), false, (b), (short)0, (c), false, false)

static __device__ __forceinline__ v16bf cat16(v8bf lo, v8bf hi) {
  return __builtin_shufflevector(lo, hi, 0,1,2,3,4,5,6,7,8,9,10,11,12,13,14,15);
}

// ---------------------------------------------------------------------------
// Tensor Data Mover: DMA a [64 x 512] bf16 tile (row stride 4096 elems) from
// global V^T into LDS. One instruction per chunk, tracked by TENSORcnt.
// ---------------------------------------------------------------------------
#if __has_builtin(__builtin_amdgcn_tensor_load_to_lds)
#define HAVE_TDM 1
typedef __attribute__((ext_vector_type(4))) unsigned int u32x4;
typedef __attribute__((ext_vector_type(4))) int          i32x4;
typedef __attribute__((ext_vector_type(8))) int          i32x8;

static __device__ __forceinline__ void tdm_load_v_chunk(const __bf16* gsrc,
                                                        unsigned lds_off) {
  unsigned long long ga = (unsigned long long)(size_t)gsrc;
  u32x4 g0;
  g0[0] = 1u;                                              // count=1 (valid user D#)
  g0[1] = lds_off;                                         // lds_addr (bytes)
  g0[2] = (unsigned)(ga & 0xffffffffu);                    // global_addr[31:0]
  g0[3] = (unsigned)((ga >> 32) & 0x1ffffffu) | (2u << 30); // addr[56:32] | type=2
  i32x8 g1;
  g1[0] = (int)(1u << 16);          // workgroup_mask=0, data_size=1 (2 bytes)
  g1[1] = (int)(4096u << 16);       // tensor_dim0 = 4096 (bits 63:48)
  g1[2] = (int)(64u << 16);         // tensor_dim0 hi = 0 | tensor_dim1 = 64
  g1[3] = (int)(512u << 16);        // tensor_dim1 hi = 0 | tile_dim0 = 512
  g1[4] = 64;                       // tile_dim1 = 64, tile_dim2 = 0
  g1[5] = 4096;                     // tensor_dim0_stride lo (4096 elems)
  g1[6] = 0;                        // stride hi, dim1_stride lo
  g1[7] = 0;
  i32x4 z4 = {0, 0, 0, 0};
#if defined(__clang_major__) && (__clang_major__ >= 23)
  i32x8 z8 = {0, 0, 0, 0, 0, 0, 0, 0};
  __builtin_amdgcn_tensor_load_to_lds(g0, g1, z4, z4, z8, 0);
#else
  __builtin_amdgcn_tensor_load_to_lds(g0, g1, z4, z4, 0);
#endif
}

static __device__ __forceinline__ void wait_tensorcnt_le1() {
#if __has_builtin(__builtin_amdgcn_s_wait_tensorcnt)
  __builtin_amdgcn_s_wait_tensorcnt(1);
#else
  asm volatile("s_wait_tensorcnt 0x1" ::: "memory");
#endif
}
static __device__ __forceinline__ void wait_tensorcnt_le0() {
#if __has_builtin(__builtin_amdgcn_s_wait_tensorcnt)
  __builtin_amdgcn_s_wait_tensorcnt(0);
#else
  asm volatile("s_wait_tensorcnt 0x0" ::: "memory");
#endif
}
#else
#define HAVE_TDM 0
#endif

// -------------------------------------------------------------------------
// Kernel 1: fused input LayerNorm + K/V projection (WMMA bf16).
// 128 rows per block. K stored row-major bf16 [b][n][d]; V stored transposed
// bf16 [b][d][n] so both feed WMMA fragments with contiguous vector loads.
// -------------------------------------------------------------------------
__global__ void __launch_bounds__(256) k_ln_kv(
    const float* __restrict__ inputs,
    const float* __restrict__ lnw, const float* __restrict__ lnb,
    const float* __restrict__ Wk,  const float* __restrict__ Wv,
    __bf16* __restrict__ Kout, __bf16* __restrict__ Vt)
{
  __shared__ __align__(16) __bf16 XKo[128 * 64];   // x_norm -> K staging -> Vt staging
  __shared__ __align__(16) __bf16 Wks[64 * 64];    // row-major [e][d]
  __shared__ __align__(16) __bf16 Wvs[64 * 64];

  const int tid = threadIdx.x;

  for (int i = tid; i < 64 * 64; i += 256) {
    Wks[i] = (__bf16)Wk[i];
    Wvs[i] = (__bf16)Wv[i];
  }

  if (tid < 128) {
    const long long grow = (long long)blockIdx.x * 128 + tid;
    const float4* rp = (const float4*)(inputs + grow * 64);
    float x[64];
    float sum = 0.f;
#pragma unroll
    for (int i = 0; i < 16; i++) {
      float4 t = rp[i];
      x[4*i+0] = t.x; x[4*i+1] = t.y; x[4*i+2] = t.z; x[4*i+3] = t.w;
      sum += t.x + t.y + t.z + t.w;
    }
    float mean = sum * (1.f / 64.f);
    float sq = 0.f;
#pragma unroll
    for (int i = 0; i < 64; i++) { float d = x[i] - mean; sq += d * d; }
    float rstd = rsqrtf(sq * (1.f / 64.f) + 1e-5f);
#pragma unroll
    for (int i = 0; i < 64; i++)
      XKo[tid * 64 + i] = (__bf16)((x[i] - mean) * rstd * lnw[i] + lnb[i]);
  }
  __syncthreads();

  const int lane = tid & 31, wave = tid >> 5;
  const int half = lane >> 4, lm = lane & 15;
  const int mt = wave;                      // 8 waves, 8 M-tiles of 16 rows

  const __bf16* arow = &XKo[(mt * 16 + lm) * 64];
  // A 16x32 fragment in this code's contiguous k order: element j -> K = half*16 + j
  v16bf a0 = cat16(*(const v8bf*)(arow + half * 16),      *(const v8bf*)(arow + 8 + half * 16));
  v16bf a1 = cat16(*(const v8bf*)(arow + 32 + half * 16), *(const v8bf*)(arow + 40 + half * 16));

  v8f cvacc[4];
#pragma unroll
  for (int et = 0; et < 4; et++) {
    // B 32x16 fragment, same k order as A: N = lm, K = half*16 + j
    const __bf16* brK = &Wks[(et * 16 + lm) * 64 + half * 16];
    v16bf bk0 = cat16(*(const v8bf*)(brK),      *(const v8bf*)(brK + 8));
    v16bf bk1 = cat16(*(const v8bf*)(brK + 32), *(const v8bf*)(brK + 40));
    v8f ck = {};
    ck = WMMA_BF16(a0, bk0, ck);
    ck = WMMA_BF16(a1, bk1, ck);

    const __bf16* brV = &Wvs[(et * 16 + lm) * 64 + half * 16];
    v16bf bv0 = cat16(*(const v8bf*)(brV),      *(const v8bf*)(brV + 8));
    v16bf bv1 = cat16(*(const v8bf*)(brV + 32), *(const v8bf*)(brV + 40));
    v8f cv = {};
    cv = WMMA_BF16(a0, bv0, cv);
    cv = WMMA_BF16(a1, bv1, cv);
    cvacc[et] = cv;

#pragma unroll
    for (int c = 0; c < 8; c++)
      XKo[(mt * 16 + c + 8 * half) * 64 + (et * 16 + lm)] = (__bf16)ck[c];
  }
  __syncthreads();

  {
    v8bf* Kg = (v8bf*)(Kout + (long long)blockIdx.x * 128 * 64);
    const v8bf* Ks = (const v8bf*)XKo;
#pragma unroll
    for (int i = 0; i < 4; i++) Kg[i * 256 + tid] = Ks[i * 256 + tid];
  }
  __syncthreads();

#pragma unroll
  for (int et = 0; et < 4; et++) {
#pragma unroll
    for (int c = 0; c < 8; c++)
      XKo[(et * 16 + lm) * 128 + (mt * 16 + c + 8 * half)] = (__bf16)cvacc[et][c];
  }
  __syncthreads();

  {
    const int bidx = blockIdx.x >> 5;          // 32 blocks per batch
    const int n0   = (blockIdx.x & 31) * 128;
    const v8bf* Vs = (const v8bf*)XKo;
#pragma unroll
    for (int i = 0; i < 4; i++) {
      int c8 = i * 256 + tid;
      int d  = c8 >> 4;
      int nl = (c8 & 15) * 8;
      *(v8bf*)(Vt + ((long long)bidx * 64 + d) * 4096 + n0 + nl) = Vs[c8];
    }
  }
}

// -------------------------------------------------------------------------
// Kernel 2: 3 slot-attention iterations, one block per batch.
// WMMA for logits (K @ q^T) and updates (attn^T @ V); V chunks staged into
// LDS by the Tensor Data Mover with double buffering; softmax / GRU / LN /
// MLP in exact fp32.
// -------------------------------------------------------------------------
struct SmemT {
  float slots[16][64];
  float sprev[16][64];
  float sn[16][64];
  float h1[16][64];
  float upd[16][64];
  float wacc[8][16][64];
  float logits[512][16];
  __bf16 qb[16][64];
  __bf16 attnT[16][512];        // transposed attention, rows 7..15 stay zero
  __bf16 vstage[2][64][512];    // TDM double buffer: [d][n_local]
  float rowm[16];
  float rowv[16];
};

__global__ void __launch_bounds__(256) k_slots(
    const __bf16* __restrict__ Kb, const __bf16* __restrict__ Vt,
    const float* __restrict__ noise,
    const float* __restrict__ ln_slots_w, const float* __restrict__ ln_slots_b,
    const float* __restrict__ ln_mlp_w,   const float* __restrict__ ln_mlp_b,
    const float* __restrict__ mu, const float* __restrict__ log_sigma,
    const float* __restrict__ Wq,
    const float* __restrict__ W_ih, const float* __restrict__ W_hh,
    const float* __restrict__ b_ih, const float* __restrict__ b_hh,
    const float* __restrict__ W1, const float* __restrict__ b1,
    const float* __restrict__ W2, const float* __restrict__ b2,
    float* __restrict__ out)
{
  extern __shared__ __align__(16) char smem_raw[];
  SmemT& sm = *(SmemT*)smem_raw;

  const int tid  = threadIdx.x;
  const int b    = blockIdx.x;
  const int lane = tid & 31, wave = tid >> 5;
  const int half = lane >> 4, lm = lane & 15;
  const float scale = 0.125f;                  // D^-0.5
  const __bf16* Vbase = Vt + (long long)b * 64 * 4096;

  // slots = mu + exp(log_sigma) * noise ; pad rows 7..15 with zeros.
  for (int idx = tid; idx < 16 * 64; idx += 256) {
    int s = idx >> 6, d = idx & 63;
    float v = 0.f;
    if (s < 7) v = mu[d] + __expf(log_sigma[d]) * noise[(b * 7 + s) * 64 + d];
    sm.slots[s][d] = v;
  }
  // zero the padded attn^T rows once (never touched again)
  for (int idx = tid; idx < 9 * 512; idx += 256)
    sm.attnT[7 + (idx >> 9)][idx & 511] = (__bf16)0.f;
  __syncthreads();

  for (int it = 0; it < 3; it++) {
    int cur = 0;
#if HAVE_TDM
    if (wave == 0)  // kick off first V chunk; overlaps with LN/q below
      tdm_load_v_chunk(Vbase, (unsigned)(size_t)&sm.vstage[0][0][0]);
#endif
    // ---- LN(slots) row stats ----
    if (tid < 7) {
      float s1 = 0.f, s2 = 0.f;
      for (int d = 0; d < 64; d++) s1 += sm.slots[tid][d];
      float m = s1 * (1.f / 64.f);
      for (int d = 0; d < 64; d++) { float dd = sm.slots[tid][d] - m; s2 += dd * dd; }
      sm.rowm[tid] = m; sm.rowv[tid] = s2 * (1.f / 64.f);
    }
    __syncthreads();
    for (int idx = tid; idx < 16 * 64; idx += 256) {
      int s = idx >> 6, d = idx & 63;
      float v = sm.slots[s][d];
      sm.sprev[s][d] = v;
      float snv = 0.f;
      if (s < 7)
        snv = (v - sm.rowm[s]) * rsqrtf(sm.rowv[s] + 1e-5f) * ln_slots_w[d] + ln_slots_b[d];
      sm.sn[s][d] = snv;
    }
    __syncthreads();
    // ---- q = LN(slots) @ Wq^T * scale, stored bf16 (rows >= 7 zero) ----
    for (int idx = tid; idx < 16 * 64; idx += 256) {
      int s = idx >> 6, e = idx & 63;
      float acc = 0.f;
      if (s < 7) {
        const float* wr = Wq + e * 64;
#pragma unroll 8
        for (int d = 0; d < 64; d++) acc += sm.sn[s][d] * wr[d];
        acc *= scale;
      }
      sm.qb[s][e] = (__bf16)acc;
    }
    __syncthreads();

    v8f acc0 = {}, acc1 = {}, acc2 = {}, acc3 = {};   // per-wave updates accumulators

    for (int cb = 0; cb < 4096; cb += 512) {
      const bool has_next = (cb + 512) < 4096;
#if HAVE_TDM
      if (wave == 0 && has_next)   // DMA next chunk into the other buffer
        tdm_load_v_chunk(Vbase + (cb + 512),
                         (unsigned)(size_t)&sm.vstage[cur ^ 1][0][0]);
#endif
      if (has_next)   // prefetch next K chunk (global_prefetch_b8)
        __builtin_prefetch(Kb + ((long long)b * 4096 + cb + 512 + wave * 64) * 64, 0, 1);

      // ---- logits = K @ q^T : 32 M-tiles, 4 per wave ----
#pragma unroll
      for (int q4 = 0; q4 < 4; q4++) {
        int mtl = wave * 4 + q4;
        const __bf16* arow = Kb + ((long long)b * 4096 + cb + mtl * 16 + lm) * 64;
        v16bf a0 = cat16(*(const v8bf*)(arow + half * 16),      *(const v8bf*)(arow + 8 + half * 16));
        v16bf a1 = cat16(*(const v8bf*)(arow + 32 + half * 16), *(const v8bf*)(arow + 40 + half * 16));
        const __bf16* brq = &sm.qb[lm][half * 16];
        v16bf q0 = cat16(*(const v8bf*)(brq),      *(const v8bf*)(brq + 8));
        v16bf q1 = cat16(*(const v8bf*)(brq + 32), *(const v8bf*)(brq + 40));
        v8f c = {};
        c = WMMA_BF16(a0, q0, c);
        c = WMMA_BF16(a1, q1, c);
#pragma unroll
        for (int cc = 0; cc < 8; cc++)
          sm.logits[mtl * 16 + cc + 8 * half][lm] = c[cc];
      }
      __syncthreads();
      // ---- softmax + eps renorm (fp32); store attn transposed as bf16 ----
      for (int r = tid; r < 512; r += 256) {
        float l[7], mx = -1e30f;
#pragma unroll
        for (int s = 0; s < 7; s++) { l[s] = sm.logits[r][s]; mx = fmaxf(mx, l[s]); }
        float sum = 0.f;
#pragma unroll
        for (int s = 0; s < 7; s++) { l[s] = __expf(l[s] - mx); sum += l[s]; }
        float inv = 1.f / sum, den = 0.f;
#pragma unroll
        for (int s = 0; s < 7; s++) { l[s] = l[s] * inv + 1e-8f; den += l[s]; }
        float dinv = 1.f / den;
#pragma unroll
        for (int s = 0; s < 7; s++) sm.attnT[s][r] = (__bf16)(l[s] * dinv);
      }
#if HAVE_TDM
      if (wave == 0) {  // TDM in-order: <=1 outstanding means chunk cb landed
        if (has_next) wait_tensorcnt_le1();
        else          wait_tensorcnt_le0();
      }
#else
      // fallback: synchronous cooperative stage of V chunk into vstage[cur]
      for (int i = 0; i < 16; i++) {
        int c8 = i * 256 + tid;               // 8-elem chunks, 4096 total
        int d  = c8 >> 6;
        int nl = (c8 & 63) * 8;
        *(v8bf*)&sm.vstage[cur][d][nl] =
            *(const v8bf*)(Vbase + (long long)d * 4096 + cb + nl);
      }
#endif
      __syncthreads();
      // ---- updates += attn^T @ V : 16 K-steps, 2 per wave, accum in regs ----
#pragma unroll
      for (int kq = 0; kq < 2; kq++) {
        int ib = (wave * 2 + kq) * 32;
        const __bf16* ar = &sm.attnT[lm][ib];
        v16bf a = cat16(*(const v8bf*)(ar + half * 16),
                        *(const v8bf*)(ar + 8 + half * 16));
        const __bf16* vb = &sm.vstage[cur][lm][ib + half * 16];
        v16bf b0 = cat16(*(const v8bf*)(vb),             *(const v8bf*)(vb + 8));
        acc0 = WMMA_BF16(a, b0, acc0);
        v16bf b1v = cat16(*(const v8bf*)(vb + 16 * 512), *(const v8bf*)(vb + 16 * 512 + 8));
        acc1 = WMMA_BF16(a, b1v, acc1);
        v16bf b2v = cat16(*(const v8bf*)(vb + 32 * 512), *(const v8bf*)(vb + 32 * 512 + 8));
        acc2 = WMMA_BF16(a, b2v, acc2);
        v16bf b3v = cat16(*(const v8bf*)(vb + 48 * 512), *(const v8bf*)(vb + 48 * 512 + 8));
        acc3 = WMMA_BF16(a, b3v, acc3);
      }
      __syncthreads();
      cur ^= 1;
    }
    // ---- cross-wave reduction of updates ----
#pragma unroll
    for (int cc = 0; cc < 8; cc++) {
      int srow = cc + 8 * half;
      sm.wacc[wave][srow][ 0 + lm] = acc0[cc];
      sm.wacc[wave][srow][16 + lm] = acc1[cc];
      sm.wacc[wave][srow][32 + lm] = acc2[cc];
      sm.wacc[wave][srow][48 + lm] = acc3[cc];
    }
    __syncthreads();
    for (int idx = tid; idx < 16 * 64; idx += 256) {
      int s = idx >> 6, d = idx & 63;
      float v = 0.f;
#pragma unroll
      for (int w = 0; w < 8; w++) v += sm.wacc[w][s][d];
      sm.upd[s][d] = v;
    }
    __syncthreads();
    // ---- GRUCell (fp32) ----
    for (int idx = tid; idx < 7 * 64; idx += 256) {
      int s = idx >> 6, d = idx & 63;
      float gir = b_ih[d], giz = b_ih[64 + d], gin = b_ih[128 + d];
      float ghr = b_hh[d], ghz = b_hh[64 + d], ghn = b_hh[128 + d];
      const float* wir = W_ih + d * 64;
      const float* wiz = W_ih + (64 + d) * 64;
      const float* win = W_ih + (128 + d) * 64;
      const float* whr = W_hh + d * 64;
      const float* whz = W_hh + (64 + d) * 64;
      const float* whn = W_hh + (128 + d) * 64;
#pragma unroll 8
      for (int e = 0; e < 64; e++) {
        float u = sm.upd[s][e], h = sm.sprev[s][e];
        gir += u * wir[e]; giz += u * wiz[e]; gin += u * win[e];
        ghr += h * whr[e]; ghz += h * whz[e]; ghn += h * whn[e];
      }
      float r = 1.f / (1.f + __expf(-(gir + ghr)));
      float z = 1.f / (1.f + __expf(-(giz + ghz)));
      float n = tanhf(gin + r * ghn);
      sm.slots[s][d] = (1.f - z) * n + z * sm.sprev[s][d];
    }
    __syncthreads();
    // ---- residual MLP (fp32) ----
    if (tid < 7) {
      float s1 = 0.f, s2 = 0.f;
      for (int d = 0; d < 64; d++) s1 += sm.slots[tid][d];
      float m = s1 * (1.f / 64.f);
      for (int d = 0; d < 64; d++) { float dd = sm.slots[tid][d] - m; s2 += dd * dd; }
      sm.rowm[tid] = m; sm.rowv[tid] = s2 * (1.f / 64.f);
    }
    __syncthreads();
    for (int idx = tid; idx < 7 * 64; idx += 256) {
      int s = idx >> 6, d = idx & 63;
      sm.sn[s][d] = (sm.slots[s][d] - sm.rowm[s]) * rsqrtf(sm.rowv[s] + 1e-5f)
                    * ln_mlp_w[d] + ln_mlp_b[d];
    }
    __syncthreads();
    for (int idx = tid; idx < 7 * 64; idx += 256) {
      int s = idx >> 6, e = idx & 63;
      float acc = b1[e];
      const float* wr = W1 + e * 64;
#pragma unroll 8
      for (int d = 0; d < 64; d++) acc += sm.sn[s][d] * wr[d];
      sm.h1[s][e] = fmaxf(acc, 0.f);
    }
    __syncthreads();
    for (int idx = tid; idx < 7 * 64; idx += 256) {
      int s = idx >> 6, d = idx & 63;
      float acc = b2[d];
      const float* wr = W2 + d * 64;
#pragma unroll 8
      for (int e = 0; e < 64; e++) acc += sm.h1[s][e] * wr[e];
      sm.slots[s][d] += acc;
    }
    __syncthreads();
  }

  for (int idx = tid; idx < 7 * 64; idx += 256)
    out[(long long)b * 448 + idx] = sm.slots[idx >> 6][idx & 63];
}

// -------------------------------------------------------------------------
extern "C" void kernel_launch(void* const* d_in, const int* in_sizes, int n_in,
                              void* d_out, int out_size, void* d_ws, size_t ws_size,
                              hipStream_t stream) {
  (void)in_sizes; (void)n_in; (void)out_size; (void)ws_size;
  const float* inputs     = (const float*)d_in[0];
  const float* noise      = (const float*)d_in[1];
  const float* ln_in_w    = (const float*)d_in[2];
  const float* ln_in_b    = (const float*)d_in[3];
  const float* ln_slots_w = (const float*)d_in[4];
  const float* ln_slots_b = (const float*)d_in[5];
  const float* ln_mlp_w   = (const float*)d_in[6];
  const float* ln_mlp_b   = (const float*)d_in[7];
  const float* mu         = (const float*)d_in[8];
  const float* log_sigma  = (const float*)d_in[9];
  const float* Wq         = (const float*)d_in[10];
  const float* Wk         = (const float*)d_in[11];
  const float* Wv         = (const float*)d_in[12];
  const float* W_ih       = (const float*)d_in[13];
  const float* W_hh       = (const float*)d_in[14];
  const float* b_ih       = (const float*)d_in[15];
  const float* b_hh       = (const float*)d_in[16];
  const float* W1         = (const float*)d_in[17];
  const float* b1         = (const float*)d_in[18];
  const float* W2         = (const float*)d_in[19];
  const float* b2         = (const float*)d_in[20];

  // Workspace: K bf16 [256][4096][64] (128 MB) + Vt bf16 [256][64][4096] (128 MB)
  __bf16* Kb = (__bf16*)d_ws;
  __bf16* Vt = Kb + (size_t)256 * 4096 * 64;

  k_ln_kv<<<8192, 256, 0, stream>>>(inputs, ln_in_w, ln_in_b, Wk, Wv, Kb, Vt);
  k_slots<<<256, 256, sizeof(SmemT), stream>>>(
      Kb, Vt, noise, ln_slots_w, ln_slots_b, ln_mlp_w, ln_mlp_b,
      mu, log_sigma, Wq, W_ih, W_hh, b_ih, b_hh, W1, b1, W2, b2,
      (float*)d_out);
}
